// ConditionalRandomField_39341900431467
// MI455X (gfx1250) — matrix operand order repaired and checked by reading, not compiled, not run on hardware
//
#include <hip/hip_runtime.h>
#include <hip/hip_bf16.h>
#include <math.h>

typedef __attribute__((ext_vector_type(16))) _Float16 v16h;
typedef __attribute__((ext_vector_type(8)))  float    v8f;

#define K_TAGS   128
#define T_LEN    512
#define B_BATCH  512
#define START_ID 0
#define END_ID   1
#define NEGV     (-10000.0f)

// ---------------------------------------------------------------------------
// Kernel 1: E[k][n] = exp(transition[n][k]) as f16 (B-fragment-friendly layout)
// ---------------------------------------------------------------------------
__global__ void crf_prep_expT(const float* __restrict__ trans,
                              _Float16* __restrict__ expT) {
  int idx = blockIdx.x * blockDim.x + threadIdx.x;
  if (idx < K_TAGS * K_TAGS) {
    int k = idx >> 7;    // reduction index i
    int n = idx & 127;   // destination tag j
    expT[idx] = (_Float16)__expf(trans[n * K_TAGS + k]);
  }
}

// ---------------------------------------------------------------------------
// Kernel 2: forward algorithm. One block = 16 batches (WMMA M), 8 waves, each
// wave owns one 16-wide N tile. Per timestep (2 barriers):
//   - async-issue emission tile t+1 into LDS (double buffered, ASYNCcnt)
//   - rowmax via width-16 shfl_xor; expA = exp(alpha - max) f16
//   - s_wait_asynccnt (tile t landed) + barrier
//   - 4x v_wmma_f32_16x16x32_f16 ; alpha = emit + max + log(acc) (masked)
// ---------------------------------------------------------------------------
__launch_bounds__(256, 1)
__global__ void crf_forward(const float* __restrict__ x,
                            const float* __restrict__ mask,
                            const float* __restrict__ trans,
                            const _Float16* __restrict__ expT,
                            float* __restrict__ out) {
  __shared__ float    alpha[16][K_TAGS];     // running log-probs (f32)
  __shared__ _Float16 expA[16][K_TAGS];      // A-matrix staging (f16)
  __shared__ float    xe[2][16][K_TAGS];     // double-buffered emission tiles
  __shared__ float    mrow[16];              // per-batch max
  __shared__ float    mk[16];                // per-batch mask at step t

  const int tid   = threadIdx.x;
  const int bBase = blockIdx.x * 16;
  const int lane  = tid & 31;
  const int ntile = tid >> 5;               // wave id == owned N tile (0..7)
  const int nLane = lane & 15;
  const int mGrp  = (lane < 16) ? 0 : 8;

  const unsigned long long xbase =
      (unsigned long long)(x + (size_t)bBase * T_LEN * K_TAGS);

  // init alpha
  for (int c = tid; c < 16 * K_TAGS; c += 256) {
    int b = c >> 7, j = c & 127;
    alpha[b][j] = (j == START_ID) ? 0.0f : NEGV;
  }

  // Loop-invariant B fragments (32x16 f16 per K-chunk):
  // table order: elem e<8 -> k = 8h + e, e>=8 -> k = 16 + 8h + (e-8); n = lane%16
  v16h bf[4];
  for (int kc = 0; kc < 4; ++kc) {
    for (int e = 0; e < 16; ++e) {
      int vg = e >> 1, hf = e & 1;
      int k  = kc * 32 + ((lane < 16) ? 0 : 8) + ((vg < 4) ? (vg * 2 + hf) : (8 + vg * 2 + hf));
      bf[kc][e] = expT[k * K_TAGS + (ntile * 16 + nLane)];
    }
  }

  // Async-issue one 16x128 f32 emission tile: 2x b128 per thread (8 KB total).
  // Per wave: exactly 2 ASYNCcnt increments per tile.
  auto issue_tile = [&](int t, int buf) {
    for (int r = 0; r < 2; ++r) {
      int idx4 = tid + r * 256;             // 16-byte chunk id, 0..511
      int b = idx4 >> 5;                    // batch row in tile
      int j = (idx4 & 31) * 4;              // starting tag column
      unsigned voff = (unsigned)(((b * T_LEN + t) * K_TAGS + j) * 4);
      unsigned lds  = (unsigned)(uintptr_t)&xe[buf][b][j];
      asm volatile("global_load_async_to_lds_b128 %0, %1, %2"
                   :: "v"(lds), "v"(voff), "s"(xbase) : "memory");
    }
  };

  issue_tile(0, 0);           // prologue: tile for t=0
  __syncthreads();

  for (int t = 0; t < T_LEN; ++t) {
    const int buf = t & 1;
    // ---- phase 1: issue t+1 tile; rowmax + expA via width-16 shuffles
    if (t + 1 < T_LEN) issue_tile(t + 1, buf ^ 1);
    {
      int b = tid >> 4, part = tid & 15;    // 16 threads per batch (half-wave)
      float av[8];
      float mx = -3.0e38f;
      for (int q = 0; q < 8; ++q) {
        av[q] = alpha[b][part * 8 + q];
        mx = fmaxf(mx, av[q]);
      }
      for (int off = 8; off >= 1; off >>= 1)
        mx = fmaxf(mx, __shfl_xor(mx, off, 16));
      if (part == 0) mrow[b] = mx;
      for (int q = 0; q < 8; ++q)
        expA[b][part * 8 + q] = (_Float16)__expf(av[q] - mx);
      if (tid < 16) mk[tid] = mask[(size_t)(bBase + tid) * T_LEN + t];
    }
    // tile t's async transfers are the oldest; <=2 outstanding => t landed
    if (t + 1 < T_LEN) asm volatile("s_wait_asynccnt 0x2" ::: "memory");
    else               asm volatile("s_wait_asynccnt 0x0" ::: "memory");
    __syncthreads();
    // ---- phase 2: WMMA + logsumexp finalize
    {
      v8f acc = {};
      const int M = lane & 15;              // A layout: both lane halves M=0..15
      for (int kc = 0; kc < 4; ++kc) {
        v16h af;
        for (int e = 0; e < 16; ++e) {
          int vg = e >> 1, hf = e & 1;
          int k = (vg < 4) ? (vg * 2 + hf + mGrp)
                           : (16 + (vg - 4) * 2 + hf + mGrp);
          af[e] = expA[M][kc * 32 + k];
        }
        acc = __builtin_amdgcn_wmma_f32_16x16x32_f16(
            false, af, false, bf[kc], (short)0, acc, false, false);
      }
      // C layout: vgpr r -> M = r (+8 for hi lanes), N = lane%16
      for (int r = 0; r < 8; ++r) {
        int Mb  = r + mGrp;
        int col = ntile * 16 + nLane;
        float newv = xe[buf][Mb][col] + mrow[Mb] + __logf(fmaxf(acc[r], 1e-30f));
        float oldv = alpha[Mb][col];
        alpha[Mb][col] = (mk[Mb] > 0.0f) ? newv : oldv;
      }
    }
    __syncthreads();
  }

  // ---- epilogue: forward_score[b] = logsumexp_j(alpha[b,j] + trans[END,j])
  if (tid < 16) {
    const float* te = trans + END_ID * K_TAGS;
    float mx = -3.0e38f;
    for (int j = 0; j < K_TAGS; ++j) mx = fmaxf(mx, alpha[tid][j] + te[j]);
    float s = 0.0f;
    for (int j = 0; j < K_TAGS; ++j) s += __expf(alpha[tid][j] + te[j] - mx);
    out[bBase + tid] = mx + __logf(s);
  }
}

// ---------------------------------------------------------------------------
// Kernel 3: gold score (gather, memory-bound); subtracts from out[].
// ---------------------------------------------------------------------------
__global__ void crf_gold(const float* __restrict__ x,
                         const int* __restrict__ tags,
                         const float* __restrict__ mask,
                         const float* __restrict__ trans,
                         float* __restrict__ out) {
  int wave = threadIdx.x >> 5;
  int lane = threadIdx.x & 31;
  int b = blockIdx.x * 8 + wave;
  if (b >= B_BATCH) return;
  const int*   tg = tags + (size_t)b * (T_LEN + 1);
  const float* mr = mask + (size_t)b * T_LEN;
  const float* xb = x + (size_t)b * T_LEN * K_TAGS;
  float score = 0.0f, len = 0.0f;
  for (int t = lane; t < T_LEN; t += 32) {
    int cur  = tg[t + 1];
    int prev = tg[t];
    float m  = mr[t];
    score += (xb[(size_t)t * K_TAGS + cur] + trans[cur * K_TAGS + prev]) * m;
    len   += m;
  }
  for (int off = 16; off >= 1; off >>= 1) {
    score += __shfl_xor(score, off, 32);
    len   += __shfl_xor(len, off, 32);
  }
  if (lane == 0) {
    int L = (int)(len + 0.5f);
    float gold = score + trans[END_ID * K_TAGS + tg[L]];
    out[b] = out[b] - gold;
  }
}

// ---------------------------------------------------------------------------
extern "C" void kernel_launch(void* const* d_in, const int* in_sizes, int n_in,
                              void* d_out, int out_size, void* d_ws, size_t ws_size,
                              hipStream_t stream) {
  (void)in_sizes; (void)n_in; (void)out_size; (void)ws_size;
  const float* x     = (const float*)d_in[0];
  const int*   tags  = (const int*)d_in[1];
  const float* mask  = (const float*)d_in[2];
  const float* trans = (const float*)d_in[3];
  float*       out   = (float*)d_out;
  _Float16*    expT  = (_Float16*)d_ws;   // 128*128*2 = 32 KB scratch

  crf_prep_expT<<<(K_TAGS * K_TAGS + 255) / 256, 256, 0, stream>>>(trans, expT);
  crf_forward<<<B_BATCH / 16, 256, 0, stream>>>(x, mask, trans, expT, out);
  crf_gold<<<B_BATCH / 8, 256, 0, stream>>>(x, tags, mask, trans, out);
}
